// Attention2D_66881230734070
// MI455X (gfx1250) — matrix operand order repaired and checked by reading, not compiled, not run on hardware
//
#include <hip/hip_runtime.h>
#include <hip/hip_bf16.h>

// ---------------- types ----------------
typedef __attribute__((ext_vector_type(16))) __bf16 v16bf;
typedef __attribute__((ext_vector_type(8)))  __bf16 v8bf;
typedef __attribute__((ext_vector_type(8)))  float  v8f;

#define CC   512
#define HW   4096
#define BB   4
#define NH   8
#define DH   64

static __device__ __forceinline__ v8f wmma_bf16(v16bf a, v16bf b, v8f c) {
    return __builtin_amdgcn_wmma_f32_16x16x32_bf16(false, a, false, b, (short)0, c,
                                                   false, false);
}

// A-fragment pattern: 16 contiguous elems at p
static __device__ __forceinline__ v16bf ldA(const __bf16* p) {
    v8bf lo = *(const v8bf*)p;
    v8bf hi = *(const v8bf*)(p + 8);
    v16bf r;
#pragma unroll
    for (int i = 0; i < 8; ++i) { r[i] = lo[i]; r[i + 8] = hi[i]; }
    return r;
}

static __device__ __forceinline__ v16bf ld16c(const __bf16* p) {  // 16 contiguous
    v8bf lo = *(const v8bf*)p;
    v8bf hi = *(const v8bf*)(p + 8);
    v16bf r;
#pragma unroll
    for (int i = 0; i < 8; ++i) { r[i] = lo[i]; r[i + 8] = hi[i]; }
    return r;
}

static __device__ __forceinline__ void zero8(v8f& v) {
#pragma unroll
    for (int i = 0; i < 8; ++i) v[i] = 0.0f;
}

// ---------------- kernel 1: weight fp32 -> bf16 ----------------
__global__ void cvt_weights(const float* __restrict__ Wqkv, const float* __restrict__ Wproj,
                            __bf16* __restrict__ Wqb, __bf16* __restrict__ Wpb) {
    const int n1 = 3 * CC * CC;
    const int n2 = CC * CC;
    for (int i = blockIdx.x * blockDim.x + threadIdx.x; i < n1 + n2;
         i += gridDim.x * blockDim.x) {
        if (i < n1) Wqb[i] = (__bf16)Wqkv[i];
        else        Wpb[i - n1] = (__bf16)Wproj[i - n1];
    }
}

// ---------------- kernel 2: LayerNorm over channels, emit bf16 [16384][512] ----
__global__ void ln_kernel(const float* __restrict__ X, const float* __restrict__ gamma,
                          const float* __restrict__ beta, __bf16* __restrict__ Xn) {
    int tid = blockIdx.x * blockDim.x + threadIdx.x;   // 0..16383  = b*4096 + p
    int b = tid >> 12, p = tid & (HW - 1);
    const float* xp = X + (size_t)b * CC * HW + p;     // coalesced over p
    float s = 0.f, s2 = 0.f;
    for (int c = 0; c < CC; ++c) {
        float v = xp[(size_t)c * HW];
        s += v; s2 += v * v;
    }
    float mu = s * (1.0f / CC);
    float var = s2 * (1.0f / CC) - mu * mu;
    float rs = rsqrtf(var + 1e-5f);
    __bf16* o = Xn + (size_t)tid * CC;
    for (int c = 0; c < CC; ++c) {
        float v = (xp[(size_t)c * HW] - mu) * rs * gamma[c] + beta[c];
        o[c] = (__bf16)v;
    }
}

// ---------------- GEMM core: one wave -> 32(M) x 64(N) tile, K=512 -----------
// Software-pipelined: next K-step fragments issued before current WMMAs.
// A: row-major [*, 512] at row m0 ; Bw: row-major [n][512] at row n0
static __device__ __forceinline__ void gemm32x64(const __bf16* __restrict__ A,
                                                 const __bf16* __restrict__ Bw,
                                                 v8f acc[8]) {
    const int lane = threadIdx.x & 31;
    const int m   = lane & 15;
    const int hi  = lane >> 4;
    const int klo = hi * 16;
    const int kb  = hi * 16;
#pragma unroll
    for (int i = 0; i < 8; ++i) zero8(acc[i]);

    v16bf a0 = ldA(A + (size_t)m * CC + klo);
    v16bf a1 = ldA(A + (size_t)(m + 16) * CC + klo);
    v16bf b[4];
#pragma unroll
    for (int nt = 0; nt < 4; ++nt)
        b[nt] = ld16c(Bw + (size_t)(nt * 16 + m) * CC + kb);

#pragma unroll
    for (int k0 = 0; k0 < CC; k0 += 32) {
        v16bf na0, na1, nb[4];
        if (k0 + 32 < CC) {      // prefetch next K-step while current WMMAs run
            na0 = ldA(A + (size_t)m * CC + (k0 + 32) + klo);
            na1 = ldA(A + (size_t)(m + 16) * CC + (k0 + 32) + klo);
#pragma unroll
            for (int nt = 0; nt < 4; ++nt)
                nb[nt] = ld16c(Bw + (size_t)(nt * 16 + m) * CC + (k0 + 32) + kb);
        }
#pragma unroll
        for (int nt = 0; nt < 4; ++nt) {
            acc[nt]     = wmma_bf16(a0, b[nt], acc[nt]);
            acc[4 + nt] = wmma_bf16(a1, b[nt], acc[4 + nt]);
        }
        if (k0 + 32 < CC) {
            a0 = na0; a1 = na1;
#pragma unroll
            for (int nt = 0; nt < 4; ++nt) b[nt] = nb[nt];
        }
    }
}

// ---------------- kernel 3: QKV GEMM + bias + head split (V transposed) ------
__global__ void qkv_gemm(const __bf16* __restrict__ Xn, const __bf16* __restrict__ Wqb,
                         const float* __restrict__ bqkv,
                         __bf16* __restrict__ Qb, __bf16* __restrict__ Kb,
                         __bf16* __restrict__ Vt) {
    const int lane = threadIdx.x & 31;
    const int m = lane & 15, hi = lane >> 4;
    int w = blockIdx.x * 8 + (threadIdx.x >> 5);       // 0..12287
    int mt = w / 24, nb = w % 24;                      // M tiles: 512, N tiles: 24
    int m0 = mt * 32, n0 = nb * 64;

    v8f acc[8];
    gemm32x64(Xn + (size_t)m0 * CC, Wqb + (size_t)n0 * CC, acc);

#pragma unroll
    for (int nt = 0; nt < 4; ++nt) {
        int n = n0 + nt * 16 + m;
        float bias = bqkv[n];
        int head = n / 192;
        int j = n - head * 192;                        // 0..63 Q, 64..127 K, 128..191 V
#pragma unroll
        for (int half = 0; half < 2; ++half) {
#pragma unroll
            for (int r = 0; r < 8; ++r) {
                int mr = m0 + 16 * half + r + 8 * hi;
                int b = mr >> 12, p = mr & (HW - 1);
                __bf16 bv = (__bf16)(acc[half * 4 + nt][r] + bias);
                size_t bh = (size_t)(b * NH + head);
                if (j < 64)       Qb[(bh * HW + p) * DH + j] = bv;
                else if (j < 128) Kb[(bh * HW + p) * DH + (j - 64)] = bv;
                else              Vt[(bh * DH + (j - 128)) * HW + p] = bv;
            }
        }
    }
}

// ---------------- kernel 4: flash attention, 1 wave = 16 queries -------------
__global__ void attn_kernel(const __bf16* __restrict__ Qb, const __bf16* __restrict__ Kb,
                            const __bf16* __restrict__ Vt, __bf16* __restrict__ Ob) {
    __shared__ __attribute__((aligned(16))) __bf16 plds[8 * 512];  // 16x32 bf16 per wave
    const int lane = threadIdx.x & 31;
    const int wslot = threadIdx.x >> 5;
    int w = blockIdx.x * 8 + wslot;                    // 0..8191
    int qt = w & 255;
    int bh = w >> 8;                                   // b*8 + h
    int q0 = qt * 16;
    const __bf16* Qp = Qb + (size_t)bh * HW * DH;
    const __bf16* Kp = Kb + (size_t)bh * HW * DH;
    const __bf16* Vp = Vt + (size_t)bh * DH * HW;
    __bf16* lp = plds + wslot * 512;

    const int m = lane & 15;
    const int hi = lane >> 4;
    const int klo = hi * 16;
    const int kb  = hi * 16;

    // preload Q A-fragments (16 x 64 -> 2 fragments along d)
    v16bf aq[2];
#pragma unroll
    for (int c2 = 0; c2 < 2; ++c2)
        aq[c2] = ldA(Qp + (size_t)(q0 + m) * DH + 32 * c2 + klo);

    float mrow[8], lrow[8];
    v8f oacc[4];
#pragma unroll
    for (int r = 0; r < 8; ++r) { mrow[r] = -1e30f; lrow[r] = 0.f; }
#pragma unroll
    for (int nt = 0; nt < 4; ++nt) zero8(oacc[nt]);

    // K fragments for first key tile: [c2] x {cols 0-15, cols 16-31}
    v16bf kf0[2], kf1[2];
#pragma unroll
    for (int c2 = 0; c2 < 2; ++c2) {
        kf0[c2] = ld16c(Kp + (size_t)m * DH + 32 * c2 + kb);
        kf1[c2] = ld16c(Kp + (size_t)(16 + m) * DH + 32 * c2 + kb);
    }

    for (int kt = 0; kt < HW / 32; ++kt) {
        const int kk0 = kt * 32;
        v8f s0, s1;
        zero8(s0); zero8(s1);
#pragma unroll
        for (int c2 = 0; c2 < 2; ++c2) {
            s0 = wmma_bf16(aq[c2], kf0[c2], s0);
            s1 = wmma_bf16(aq[c2], kf1[c2], s1);
        }
        // issue V fragments for this tile and K fragments for the next tile now;
        // their latency hides under the softmax VALU block below.
        v16bf vf[4];
#pragma unroll
        for (int nt = 0; nt < 4; ++nt)
            vf[nt] = ld16c(Vp + (size_t)(nt * 16 + m) * HW + kk0 + kb);
        if (kt + 1 < HW / 32) {
            const int kk1 = kk0 + 32;
#pragma unroll
            for (int c2 = 0; c2 < 2; ++c2) {
                kf0[c2] = ld16c(Kp + (size_t)(kk1 + m) * DH + 32 * c2 + kb);
                kf1[c2] = ld16c(Kp + (size_t)(kk1 + 16 + m) * DH + 32 * c2 + kb);
            }
        }
        // online softmax per row (row = one VGPR across a 16-lane half)
#pragma unroll
        for (int r = 0; r < 8; ++r) {
            float v0 = s0[r] * 0.125f;                 // 1/sqrt(64)
            float v1 = s1[r] * 0.125f;
            float mx = fmaxf(v0, v1);
            mx = fmaxf(mx, __shfl_xor(mx, 1, 16));
            mx = fmaxf(mx, __shfl_xor(mx, 2, 16));
            mx = fmaxf(mx, __shfl_xor(mx, 4, 16));
            mx = fmaxf(mx, __shfl_xor(mx, 8, 16));
            float mnew = fmaxf(mrow[r], mx);
            float alpha = __expf(mrow[r] - mnew);
            float p0 = __expf(v0 - mnew);
            float p1 = __expf(v1 - mnew);
            float rs = p0 + p1;
            rs += __shfl_xor(rs, 1, 16);
            rs += __shfl_xor(rs, 2, 16);
            rs += __shfl_xor(rs, 4, 16);
            rs += __shfl_xor(rs, 8, 16);
            lrow[r] = lrow[r] * alpha + rs;
            mrow[r] = mnew;
#pragma unroll
            for (int nt = 0; nt < 4; ++nt) oacc[nt][r] *= alpha;
            s0[r] = p0; s1[r] = p1;
        }
        // stage P (16x32 bf16) through wave-private LDS to A-fragment layout
#pragma unroll
        for (int r = 0; r < 8; ++r) {
            int row = r + 8 * hi;
            lp[row * 32 + m]      = (__bf16)s0[r];
            lp[row * 32 + 16 + m] = (__bf16)s1[r];
        }
        asm volatile("s_wait_dscnt 0x0" ::: "memory");
        v16bf ap = ldA(lp + m * 32 + klo);
        // O += P * V  (V transposed: contiguous keys per d-column)
#pragma unroll
        for (int nt = 0; nt < 4; ++nt)
            oacc[nt] = wmma_bf16(ap, vf[nt], oacc[nt]);
    }
    // normalize and store O as bf16 [b*4096+q][head*64+d]
    float inv[8];
#pragma unroll
    for (int r = 0; r < 8; ++r) inv[r] = 1.0f / lrow[r];
    int b = bh >> 3, h = bh & 7;
#pragma unroll
    for (int nt = 0; nt < 4; ++nt) {
        int col = h * DH + nt * 16 + m;
#pragma unroll
        for (int r = 0; r < 8; ++r) {
            size_t row = (size_t)b * HW + q0 + r + 8 * hi;
            Ob[row * CC + col] = (__bf16)(oacc[nt][r] * inv[r]);
        }
    }
}

// ---------------- kernel 5: proj GEMM + bias, output (B,C,H,W) fp32 ----------
__global__ void proj_gemm(const __bf16* __restrict__ Ob, const __bf16* __restrict__ Wpb,
                          const float* __restrict__ bproj, float* __restrict__ out) {
    const int lane = threadIdx.x & 31;
    const int m = lane & 15, hi = lane >> 4;
    int w = blockIdx.x * 8 + (threadIdx.x >> 5);       // 0..4095
    int mt = w >> 3, nb = w & 7;                       // M tiles: 512, N tiles: 8
    int m0 = mt * 32, n0 = nb * 64;

    v8f acc[8];
    gemm32x64(Ob + (size_t)m0 * CC, Wpb + (size_t)n0 * CC, acc);

#pragma unroll
    for (int nt = 0; nt < 4; ++nt) {
        int n = n0 + nt * 16 + m;
        float bias = bproj[n];
#pragma unroll
        for (int half = 0; half < 2; ++half) {
#pragma unroll
            for (int r = 0; r < 8; ++r) {
                int mr = m0 + 16 * half + r + 8 * hi;
                int b = mr >> 12, p = mr & (HW - 1);
                out[((size_t)b * CC + n) * HW + p] = acc[half * 4 + nt][r] + bias;
            }
        }
    }
}

// ---------------- launcher ----------------
extern "C" void kernel_launch(void* const* d_in, const int* in_sizes, int n_in,
                              void* d_out, int out_size, void* d_ws, size_t ws_size,
                              hipStream_t stream) {
    const float* X     = (const float*)d_in[0];
    const float* gamma = (const float*)d_in[1];
    const float* beta  = (const float*)d_in[2];
    const float* Wqkv  = (const float*)d_in[3];
    const float* bqkv  = (const float*)d_in[4];
    const float* Wproj = (const float*)d_in[5];
    const float* bproj = (const float*)d_in[6];
    float* out = (float*)d_out;

    char* ws = (char*)d_ws;
    const size_t SEG = (size_t)16 * 1024 * 1024;       // 16 MB segments
    __bf16* Xn  = (__bf16*)(ws + 0 * SEG);             // [16384][512]
    __bf16* Qb  = (__bf16*)(ws + 1 * SEG);             // [32][4096][64]
    __bf16* Kb  = (__bf16*)(ws + 2 * SEG);             // [32][4096][64]
    __bf16* Vt  = (__bf16*)(ws + 3 * SEG);             // [32][64][4096]
    __bf16* Ob  = (__bf16*)(ws + 4 * SEG);             // [16384][512]
    __bf16* Wqb = (__bf16*)(ws + 5 * SEG);             // [1536][512]
    __bf16* Wpb = (__bf16*)(ws + 5 * SEG + 3 * CC * CC * sizeof(__bf16));

    cvt_weights<<<1024, 256, 0, stream>>>(Wqkv, Wproj, Wqb, Wpb);
    ln_kernel<<<64, 256, 0, stream>>>(X, gamma, beta, Xn);
    qkv_gemm<<<1536, 256, 0, stream>>>(Xn, Wqb, bqkv, Qb, Kb, Vt);   // 12288 waves
    attn_kernel<<<1024, 256, 0, stream>>>(Qb, Kb, Vt, Ob);           // 8192 waves
    proj_gemm<<<512, 256, 0, stream>>>(Ob, Wpb, bproj, out);         // 4096 waves
}
